// PixelCNN_61452392071744
// MI455X (gfx1250) — compile-verified, matrix operand order checked by reading, not run on hardware
//
#include <hip/hip_runtime.h>
#include <hip/hip_bf16.h>

typedef __attribute__((ext_vector_type(16))) _Float16 v16h;
typedef __attribute__((ext_vector_type(8)))  _Float16 v8h;
typedef __attribute__((ext_vector_type(4)))  _Float16 v4h;
typedef __attribute__((ext_vector_type(8)))  float    v8f;
typedef __attribute__((ext_vector_type(4)))  float    v4f;

#define B   8
#define HW  144          // 12*12
#define NPOS 1152        // B*HW
#define HID 32
#define NTAP_B 25        // mask-B nonzero taps
#define NTAP_A 24        // mask-A nonzero taps
#define WB_LAYER (NTAP_B * 2 * 32 * 16)   // 25600 halfs per layer
#define WB_TOTAL (3 * WB_LAYER)           // 76800 halfs
#define W0M_TOTAL (2 * NTAP_A * HID)      // 1536 floats

// branch-free tap decode, exact for t in [0,28): dy=t/7, dx=t%7
__device__ __forceinline__ void tap_dydx(int t, int& dy, int& dx) {
    dy = (t * 37) >> 8;
    dx = t - dy * 7;
}

// ---------------------------------------------------------------------------
// Prep: mask + repack weights.
//  wB  [l][t][nt][lane][j] f16 : B-fragment layout for v_wmma_f32_16x16x32_f16
//  w0m [c][t][o] f32 : mask-A w0, per input channel (0=act, 1=buf)
// ---------------------------------------------------------------------------
__global__ void pixelcnn_prep(const float* __restrict__ w0,
                              const float* __restrict__ w1,
                              const float* __restrict__ w2,
                              const float* __restrict__ w3,
                              _Float16* __restrict__ wB,
                              float* __restrict__ w0m) {
    int idx = blockIdx.x * blockDim.x + threadIdx.x;
    if (idx < WB_TOTAL) {
        int j   = idx & 15;
        int L   = (idx >> 4) & 31;
        int nt  = (idx >> 9) & 1;
        int t   = (idx >> 10) % NTAP_B;
        int l   = idx / WB_LAYER;
        int dy, dx; tap_dydx(t, dy, dx);
        int cout = nt * 16 + (L & 15);
        int kk   = ((j < 8) ? j : (j + 8)) + ((L >= 16) ? 8 : 0);
        const float* w = (l == 0) ? w1 : ((l == 1) ? w2 : w3);
        wB[idx] = (_Float16)w[(cout * HID + kk) * 49 + dy * 7 + dx];
    }
    int idx2 = idx - WB_TOTAL;
    if (idx2 >= 0 && idx2 < W0M_TOTAL) {
        int o = idx2 & 31;
        int t = (idx2 >> 5) % NTAP_A;
        int c = idx2 / (NTAP_A * HID);
        int dy, dx; tap_dydx(t, dy, dx);
        w0m[idx2] = w0[(o * 2 + c) * 49 + dy * 7 + dx];
    }
}

// ---------------------------------------------------------------------------
// One mask-B 7x7 conv layer: tap-outer / 9-jobs-inner.
// Per tap: one (double-buffered) B-fragment load shared by 9 independent
// WMMA chains. OOB handled by address-select to a zeroed LDS pad --
// unconditional ds_load_b128, no EXEC-mask control flow in the hot loop.
// ---------------------------------------------------------------------------
__device__ __forceinline__ void conv_layer_wmma(const _Float16* __restrict__ hin,
                                                _Float16* __restrict__ hout,
                                                const float* __restrict__ bias,
                                                const _Float16* __restrict__ wBl,
                                                const _Float16* __restrict__ zpad0,
                                                int wave, int lane) {
    const bool hiHalf = lane >= 16;
    const int  n      = lane & 15;
    const int  nt     = wave & 1;                  // job parity fixed per wave
    const int  cout   = nt * 16 + n;
    const float bv    = bias[cout];
    const _Float16* bptr = wBl + nt * 512 + lane * 16;   // + t*1024 per tap
    const int  c0h    = hiHalf ? 8 : 0;            // lane's K-chunk offset (halfs)
    const _Float16* zp = zpad0 + c0h;

    // per-job A-row coordinates for this lane (job = wave + 16*jj)
    int py[9], px[9];
    const _Float16* abase[9];
    #pragma unroll
    for (int jj = 0; jj < 9; ++jj) {
        const int mtile = (wave >> 1) + jj * 8;
        const int m  = mtile * 16 + n;
        const int bb = m / HW;
        const int rr = m - bb * HW;
        py[jj] = rr / 12;
        px[jj] = rr - py[jj] * 12;
        abase[jj] = hin + (bb * HW + py[jj] * 12 + px[jj]) * HID + c0h;
    }
    v8f acc[9];
    #pragma unroll
    for (int jj = 0; jj < 9; ++jj) {
        #pragma unroll
        for (int r = 0; r < 8; ++r) acc[jj][r] = bv;
    }

    v16h bcur = *(const v16h*)bptr;                // prologue load, tap 0
    #pragma unroll 1
    for (int t = 0; t < NTAP_B; ++t) {
        const int tn = (t < NTAP_B - 1) ? (t + 1) : t;
        const v16h bnext = *(const v16h*)(bptr + tn * 1024);   // prefetch
        int dy, dx; tap_dydx(t, dy, dx);
        const int sOff = ((dy - 3) * 12 + (dx - 3)) * HID;     // uniform offset
        #pragma unroll
        for (int jj = 0; jj < 9; ++jj) {
            const unsigned sy = (unsigned)(py[jj] + dy - 3);
            const unsigned sx = (unsigned)(px[jj] + dx - 3);
            const bool valid  = (sy < 12u) & (sx < 12u);
            const _Float16* ap = valid ? (abase[jj] + sOff) : zp;  // v_cndmask
            v8h lo = *(const v8h*)(ap);            // K 0..7  / 8..15
            v8h hi = *(const v8h*)(ap + 16);       // K 16..23 / 24..31
            v16h a;
            #pragma unroll
            for (int q = 0; q < 8; ++q) { a[q] = lo[q]; a[q + 8] = hi[q]; }
            acc[jj] = __builtin_amdgcn_wmma_f32_16x16x32_f16(
                false, a, false, bcur, (short)0, acc[jj], false, false);
        }
        bcur = bnext;
    }

    // C/D layout: VGPR r -> row (r | r+8), col = lane&15
    #pragma unroll
    for (int jj = 0; jj < 9; ++jj) {
        const int mtile = (wave >> 1) + jj * 8;
        #pragma unroll
        for (int r = 0; r < 8; ++r) {
            const int mm = mtile * 16 + r + (hiHalf ? 8 : 0);
            float v = acc[jj][r];
            hout[mm * HID + cout] = (_Float16)(v > 0.0f ? v : 0.0f);
        }
    }
}

// ---------------------------------------------------------------------------
// Persistent kernel: one workgroup, all 144 raster steps, all state in LDS.
// ---------------------------------------------------------------------------
__global__ __launch_bounds__(512)
void pixelcnn_persistent(const float* __restrict__ x,
                         const float* __restrict__ w0m_g,
                         const _Float16* __restrict__ wB,
                         const float* __restrict__ b0,
                         const float* __restrict__ b1,
                         const float* __restrict__ b2,
                         const float* __restrict__ b3,
                         const float* __restrict__ wo,
                         const float* __restrict__ bo,
                         float* __restrict__ out) {
    __shared__ float    h0[NPOS * HID];      // layer-0 pre-activation, f32 (incr. updated)
    __shared__ _Float16 hA[NPOS * HID];      // ping
    __shared__ _Float16 hB[NPOS * HID];      // pong
    __shared__ float    actL[NPOS];          // sigmoid(x), iteration-invariant
    __shared__ float    logitL[NPOS];
    __shared__ float    w0mL[W0M_TOTAL];
    __shared__ float    woL[HID + 1];        // wo[0..31], bo
    __shared__ __align__(64) _Float16 zpadL[32];  // 64B of zeros for OOB A-reads

    const int tid  = threadIdx.x;
    const int wave = __builtin_amdgcn_readfirstlane(tid >> 5);  // wave-uniform SGPR
    const int lane = tid & 31;

    for (int p = tid; p < NPOS; p += 512)
        actL[p] = 1.0f / (1.0f + __expf(-x[p]));
    for (int i = tid; i < W0M_TOTAL; i += 512)
        w0mL[i] = w0m_g[i];
    if (tid < HID) woL[tid] = wo[tid];
    if (tid == HID) woL[HID] = bo[0];
    if (tid < 32) zpadL[tid] = (_Float16)0.0f;
    __syncthreads();

    // h0 base = b0 + maskA-conv(act)   (buf == 0 initially)
    for (int idx = tid; idx < NPOS * HID; idx += 512) {
        const int p = idx >> 5, o = idx & 31;
        const int b = p / HW, rr = p - b * HW;
        const int py = rr / 12, px = rr - py * 12;
        float s = b0[o];
        #pragma unroll 1
        for (int t = 0; t < NTAP_A; ++t) {
            int dy, dx; tap_dydx(t, dy, dx);
            const int sy = py + dy - 3, sx = px + dx - 3;
            if ((unsigned)sy < 12u && (unsigned)sx < 12u)
                s = fmaf(w0mL[t * HID + o], actL[b * HW + sy * 12 + sx], s);
        }
        h0[idx] = s;
    }
    __syncthreads();

    #pragma unroll 1
    for (int it = 0; it < HW; ++it) {
        const int wi = it / 12, hi2 = it % 12;

        // ReLU(h0) -> hA (f16), 4 elements per thread-step (b128 in, b64 out)
        for (int i4 = tid; i4 < (NPOS * HID) / 4; i4 += 512) {
            v4f v = ((const v4f*)h0)[i4];
            v4h h;
            #pragma unroll
            for (int q = 0; q < 4; ++q)
                h[q] = (_Float16)(v[q] > 0.0f ? v[q] : 0.0f);
            ((v4h*)hA)[i4] = h;
        }
        __syncthreads();

        conv_layer_wmma(hA, hB, b1, wB + 0 * WB_LAYER, zpadL, wave, lane);
        __syncthreads();
        conv_layer_wmma(hB, hA, b2, wB + 1 * WB_LAYER, zpadL, wave, lane);
        __syncthreads();
        conv_layer_wmma(hA, hB, b3, wB + 2 * WB_LAYER, zpadL, wave, lane);
        __syncthreads();

        // 1x1 conv -> logits (vector reads of hB)
        for (int p = tid; p < NPOS; p += 512) {
            float s = woL[HID];
            const v8h* hp = (const v8h*)(hB + p * HID);
            #pragma unroll
            for (int g = 0; g < 4; ++g) {
                v8h hv = hp[g];
                #pragma unroll
                for (int q = 0; q < 8; ++q)
                    s = fmaf(woL[g * 8 + q], (float)hv[q], s);
            }
            logitL[p] = s;
        }
        __syncthreads();

        // commit pixel (wi,hi2): buf delta = sigmoid(logit); rank-1 update of h0
        for (int idx = tid; idx < NTAP_A * B * HID; idx += 512) {
            const int t = idx >> 8;            // /(8*32)
            const int b = (idx >> 5) & 7;
            const int o = idx & 31;
            int dy, dx; tap_dydx(t, dy, dx);
            const int y  = wi + 3 - dy;        // >= wi (causal)
            const int xx = hi2 + 3 - dx;
            if (y < 12 && (unsigned)xx < 12u) {
                const float lg  = logitL[b * HW + wi * 12 + hi2];
                const float dlt = 1.0f / (1.0f + __expf(-lg));
                h0[(b * HW + y * 12 + xx) * HID + o] =
                    fmaf(w0mL[NTAP_A * HID + t * HID + o], dlt,
                         h0[(b * HW + y * 12 + xx) * HID + o]);
            }
        }
        __syncthreads();
    }

    for (int p = tid; p < NPOS; p += 512) out[p] = logitL[p];
}

extern "C" void kernel_launch(void* const* d_in, const int* in_sizes, int n_in,
                              void* d_out, int out_size, void* d_ws, size_t ws_size,
                              hipStream_t stream) {
    const float* x  = (const float*)d_in[0];
    const float* w0 = (const float*)d_in[1];
    const float* b0 = (const float*)d_in[2];
    const float* w1 = (const float*)d_in[3];
    const float* b1 = (const float*)d_in[4];
    const float* w2 = (const float*)d_in[5];
    const float* b2 = (const float*)d_in[6];
    const float* w3 = (const float*)d_in[7];
    const float* b3 = (const float*)d_in[8];
    const float* wo = (const float*)d_in[9];
    const float* bo = (const float*)d_in[10];

    _Float16* wB  = (_Float16*)d_ws;                              // 76800 halfs
    float*    w0m = (float*)((char*)d_ws + WB_TOTAL * sizeof(_Float16));

    const int prep_n = WB_TOTAL + W0M_TOTAL;                      // 78336
    pixelcnn_prep<<<(prep_n + 255) / 256, 256, 0, stream>>>(w0, w1, w2, w3, wB, w0m);

    pixelcnn_persistent<<<1, 512, 0, stream>>>(
        x, w0m, wB, b0, b1, b2, b3, wo, bo, (float*)d_out);
}
